// VisionTransformerBlock_51539608065
// MI455X (gfx1250) — matrix operand order repaired and checked by reading, not compiled, not run on hardware
//
#include <hip/hip_runtime.h>
#include <hip/hip_bf16.h>
#include <math.h>

// Problem dims (compile-time)
#define B_    128
#define S_    256
#define D_    128
#define H_    8
#define HD_   16
#define MLPH_ 64
// 1/SCALE where SCALE = sqrt(D/H) = 4.0
#define INV_SCALE 0.25f

typedef __attribute__((ext_vector_type(16))) _Float16 v16h;
typedef __attribute__((ext_vector_type(8)))  float    v8f;

// ---------------------------------------------------------------------------
// CDNA5 wave32 WMMA operand packing (per ISA 7.12.2):
//  A (16x32 f16): lane L<16 -> M=L,   v0..3: K=0..7,  v4..7: K=16..23
//                 lane L>=16 -> M=L-16, v0..3: K=8..15, v4..7: K=24..31
//  B (32x16 f16): N=lane%16, hw K order as A; code packs A and B alike: lanes 0-15 K=0..15, 16-31 K=16..31
//  C/D (16x16 f32): VGPR e, lanes 0-15: M=e, N=lane; lanes 16-31: M=e+8, N=lane-16
// ---------------------------------------------------------------------------

// A operand from row-major f16 tile: element (m,k) at src[m*ld + k], K=32
static __device__ __forceinline__ v16h pack_a_h(const _Float16* src, int ld) {
  const int lane = threadIdx.x & 31;
  const int m = lane & 15, hi = lane >> 4;
  v16h a;
#pragma unroll
  for (int j = 0; j < 8; ++j) {
    const int k = (hi << 4) + (j << 1);
    a[2 * j]     = src[m * ld + k];
    a[2 * j + 1] = src[m * ld + k + 1];
  }
  return a;
}

// A operand, valid K only 0..15 (upper half zero-padded)
static __device__ __forceinline__ v16h pack_a_k16(const _Float16* src, int ld) {
  const int lane = threadIdx.x & 31;
  const int m = lane & 15, hi = lane >> 4;
  v16h a = {};
#pragma unroll
  for (int j = 0; j < 8 && !hi; ++j) {
    const int k = (j << 1);          // lanes 0-15 hold K=0..15 contiguous
    a[2 * j]     = src[m * ld + k];
    a[2 * j + 1] = src[m * ld + k + 1];
  }
  return a;
}

// B operand from row-major f16 tile: element (k,n) at src[k*ld + n], K=32
static __device__ __forceinline__ v16h pack_b_h(const _Float16* src, int ld) {
  const int lane = threadIdx.x & 31;
  const int n = lane & 15, hi = lane >> 4;
  v16h b;
#pragma unroll
  for (int j = 0; j < 8; ++j) {
    const int k = (hi << 4) + (j << 1);
    b[2 * j]     = src[k * ld + n];
    b[2 * j + 1] = src[(k + 1) * ld + n];
  }
  return b;
}

// B operand from row-major f32 weights (convert to f16): element (k,n) = src[k*ld+n]
static __device__ __forceinline__ v16h pack_b_f32(const float* src, int ld) {
  const int lane = threadIdx.x & 31;
  const int n = lane & 15, hi = lane >> 4;
  v16h b;
#pragma unroll
  for (int j = 0; j < 8; ++j) {
    const int k = (hi << 4) + (j << 1);
    b[2 * j]     = (_Float16)src[k * ld + n];
    b[2 * j + 1] = (_Float16)src[(k + 1) * ld + n];
  }
  return b;
}

// B operand = transpose of a row-major [seq][hd] tile, K (=hd) only 0..15 valid.
// element B[kd][n] = src[n*ld + kd] -> 16 contiguous f16 per active lane.
static __device__ __forceinline__ v16h pack_bT_k16(const _Float16* src, int ld) {
  const int lane = threadIdx.x & 31;
  const int n = lane & 15, hi = lane >> 4;
  v16h b = {};
  if (!hi) {
#pragma unroll
    for (int j = 0; j < 8; ++j) {
      b[2 * j]     = src[n * ld + 2 * j];
      b[2 * j + 1] = src[n * ld + 2 * j + 1];
    }
  }
  return b;
}

static __device__ __forceinline__ v8f wmma16(v16h a, v16h b, v8f c) {
  return __builtin_amdgcn_wmma_f32_16x16x32_f16(false, a, false, b, (short)0, c,
                                                false, false);
}

// ---------------------------------------------------------------------------
// K1: LayerNorm(x) -> z (f16 in LDS), then z @ {Wq,Wk,Wv} + bias, stored f16
//     as [B,H,S,HD]. 2048 blocks x 128 threads (4 waves, 6 output tiles each).
// ---------------------------------------------------------------------------
__global__ void ln_qkv_kernel(const float* __restrict__ x,
                              const float* __restrict__ wq, const float* __restrict__ bq,
                              const float* __restrict__ wk, const float* __restrict__ bk,
                              const float* __restrict__ wv, const float* __restrict__ bv,
                              const float* __restrict__ g1, const float* __restrict__ be1,
                              _Float16* __restrict__ qh, _Float16* __restrict__ kh,
                              _Float16* __restrict__ vh) {
  __shared__ _Float16 zt[16][D_];
  __shared__ float rs[16][8];
  __shared__ float rss[16][8];

  const int tid = threadIdx.x;      // 0..127
  const int row = tid >> 3;         // 0..15
  const int seg = tid & 7;          // 0..7  (16 cols each)
  const long grow = (long)blockIdx.x * 16 + row;

  // --- LayerNorm (fp32 stats) ---
  float vloc[16];
  float s = 0.f, ss = 0.f;
  const float* xr = x + grow * D_ + seg * 16;
#pragma unroll
  for (int i = 0; i < 16; ++i) {
    vloc[i] = xr[i];
    s += vloc[i];
    ss += vloc[i] * vloc[i];
  }
  rs[row][seg] = s;
  rss[row][seg] = ss;
  __syncthreads();
  float tot = 0.f, tots = 0.f;
#pragma unroll
  for (int j = 0; j < 8; ++j) { tot += rs[row][j]; tots += rss[row][j]; }
  const float mu = tot * (1.0f / D_);
  const float var = tots * (1.0f / D_) - mu * mu;
  const float rstd = rsqrtf(var + 1e-5f);
#pragma unroll
  for (int i = 0; i < 16; ++i) {
    const int col = seg * 16 + i;
    zt[row][col] = (_Float16)((vloc[i] - mu) * rstd * g1[col] + be1[col]);
  }
  __syncthreads();

  // --- QKV projection: 24 16x16 tiles, 4 waves x 6 ---
  const int wave = tid >> 5;
  const int lane = tid & 31;
  const int nlane = lane & 15, hi = lane >> 4;
#pragma unroll
  for (int t = 0; t < 6; ++t) {
    const int tile  = wave * 6 + t;      // 0..23
    const int which = tile >> 3;         // 0=q 1=k 2=v
    const int nt    = tile & 7;          // head index (16 cols per head)
    const float* W  = which == 0 ? wq : (which == 1 ? wk : wv);
    const float* Bv = which == 0 ? bq : (which == 1 ? bk : bv);
    const float bias = Bv[nt * 16 + nlane];
    v8f c;
#pragma unroll
    for (int e = 0; e < 8; ++e) c[e] = bias;

    for (int ks = 0; ks < D_; ks += 32) {
      if (ks + 32 < D_) __builtin_prefetch(W + (ks + 32) * D_ + nt * 16, 0, 1);
      v16h a = pack_a_h(&zt[0][0] + ks, D_);
      v16h b = pack_b_f32(W + ks * D_ + nt * 16, D_);
      c = wmma16(a, b, c);
    }

    _Float16* dst = which == 0 ? qh : (which == 1 ? kh : vh);
#pragma unroll
    for (int e = 0; e < 8; ++e) {
      const int m = e + hi * 8;
      const int r = blockIdx.x * 16 + m;   // global row in [0, B*S)
      const int bb = r >> 8;               // / S_
      const int si = r & (S_ - 1);
      dst[((bb * H_ + nt) * S_ + si) * HD_ + nlane] = (_Float16)c[e];
    }
  }
}

// ---------------------------------------------------------------------------
// K2: softmax-over-BATCH statistics. For each (h, q, k): running max & sumexp
//     across b=0..127 (online). One wave per (h, qtile, ktile). 2048 blocks.
// ---------------------------------------------------------------------------
__global__ void attn_stats_kernel(const _Float16* __restrict__ qh,
                                  const _Float16* __restrict__ kh,
                                  float* __restrict__ m_ws, float* __restrict__ l_ws) {
  const int idx = blockIdx.x;
  const int kt = idx & 15;
  const int qt = (idx >> 4) & 15;
  const int h  = idx >> 8;
  const int lane = threadIdx.x & 31;
  const int nlane = lane & 15, hi = lane >> 4;

  float mr[8], lr[8];
#pragma unroll
  for (int e = 0; e < 8; ++e) { mr[e] = -1e30f; lr[e] = 0.f; }

  for (int b = 0; b < B_; ++b) {
    const _Float16* qp = qh + (((b * H_ + h) * S_) + qt * 16) * HD_;
    const _Float16* kp = kh + (((b * H_ + h) * S_) + kt * 16) * HD_;
    v16h a  = pack_a_k16(qp, HD_);
    v16h bm = pack_bT_k16(kp, HD_);
    v8f c = {};
    c = wmma16(a, bm, c);
#pragma unroll
    for (int e = 0; e < 8; ++e) {
      const float sc = c[e] * INV_SCALE;
      const float mn = fmaxf(mr[e], sc);
      lr[e] = lr[e] * __expf(mr[e] - mn) + __expf(sc - mn);
      mr[e] = mn;
    }
  }
#pragma unroll
  for (int e = 0; e < 8; ++e) {
    const int m = e + hi * 8;
    const int off = ((h * S_ + qt * 16 + m) * S_) + kt * 16 + nlane;
    m_ws[off] = mr[e];
    l_ws[off] = lr[e];
  }
}

// ---------------------------------------------------------------------------
// K3: recompute scores, normalize with (m,l), attn @ V, residual add.
//     One wave per (b, h, qtile). Processes two k-tiles per K=32 WMMA.
//     Writes x1 = x + ret to d_out (fp32) and to x1h (f16) for the MLP.
// ---------------------------------------------------------------------------
__global__ void attn_apply_kernel(const _Float16* __restrict__ qh,
                                  const _Float16* __restrict__ kh,
                                  const _Float16* __restrict__ vh,
                                  const float* __restrict__ m_ws,
                                  const float* __restrict__ l_ws,
                                  const float* __restrict__ x,
                                  float* __restrict__ xout,
                                  _Float16* __restrict__ x1h) {
  __shared__ _Float16 attn_lds[16 * 32];
  const int idx = blockIdx.x;
  const int qt = idx & 15;
  const int h  = (idx >> 4) & 7;
  const int b  = idx >> 7;
  const int lane = threadIdx.x & 31;
  const int nlane = lane & 15, hi = lane >> 4;

  const _Float16* qp = qh + (((b * H_ + h) * S_) + qt * 16) * HD_;
  const v16h aQ = pack_a_k16(qp, HD_);
  v8f acc = {};

  for (int kt = 0; kt < 16; kt += 2) {
#pragma unroll
    for (int half = 0; half < 2; ++half) {
      const int ktt = kt + half;
      const _Float16* kp = kh + (((b * H_ + h) * S_) + ktt * 16) * HD_;
      v16h bK = pack_bT_k16(kp, HD_);
      v8f c = {};
      c = wmma16(aQ, bK, c);
#pragma unroll
      for (int e = 0; e < 8; ++e) {
        const int m = e + hi * 8;
        const int off = ((h * S_ + qt * 16 + m) * S_) + ktt * 16 + nlane;
        const float sc = c[e] * INV_SCALE;
        const float w = __expf(sc - m_ws[off]) / l_ws[off];
        attn_lds[m * 32 + half * 16 + nlane] = (_Float16)w;
      }
    }
    __syncthreads();  // single wave: LDS same-wave ordering + compiler waits
    v16h aW = pack_a_h(attn_lds, 32);
    const _Float16* vp = vh + (((b * H_ + h) * S_) + kt * 16) * HD_;
    v16h bV = pack_b_h(vp, HD_);   // 32 seq rows x 16 hd, contiguous tiles
    acc = wmma16(aW, bV, acc);
    __syncthreads();
  }

  // residual: x1 = x + ret
#pragma unroll
  for (int e = 0; e < 8; ++e) {
    const int m = e + hi * 8;
    const int srow = qt * 16 + m;
    const int d = h * 16 + nlane;
    const int off = (b * S_ + srow) * D_ + d;
    const float x1 = x[off] + acc[e];
    xout[off] = x1;
    x1h[off] = (_Float16)x1;
  }
}

// ---------------------------------------------------------------------------
// K4: MLP: h = gelu(x1 @ w1 + b1) (exact erf), z = h @ w2 + b2, out = x1 + z.
//     2048 blocks x 128 threads (4 waves). LN2 is computed-and-discarded in
//     the reference, so it is simply omitted.
// ---------------------------------------------------------------------------
__global__ void mlp_kernel(const _Float16* __restrict__ x1h,
                           const float* __restrict__ w1, const float* __restrict__ b1,
                           const float* __restrict__ w2, const float* __restrict__ b2,
                           float* __restrict__ out) {
  __shared__ _Float16 ht[16][MLPH_];
  const int tid = threadIdx.x;
  const int wave = tid >> 5;
  const int lane = tid & 31;
  const int nlane = lane & 15, hi = lane >> 4;
  const long rowbase = (long)blockIdx.x * 16;

  // --- phase 1: [16,128] @ [128,64] + b1, GELU -> LDS f16 ---
  {
    const int nt = wave;  // 4 n-tiles of 16
    const float bias = b1[nt * 16 + nlane];
    v8f c;
#pragma unroll
    for (int e = 0; e < 8; ++e) c[e] = bias;
    for (int ks = 0; ks < D_; ks += 32) {
      v16h a = pack_a_h(x1h + rowbase * D_ + ks, D_);
      v16h b = pack_b_f32(w1 + ks * MLPH_ + nt * 16, MLPH_);
      c = wmma16(a, b, c);
    }
#pragma unroll
    for (int e = 0; e < 8; ++e) {
      const int m = e + hi * 8;
      const float u = c[e];
      const float g = 0.5f * u * (1.0f + erff(u * 0.70710678118654752f));
      ht[m][nt * 16 + nlane] = (_Float16)g;
    }
  }
  __syncthreads();

  // --- phase 2: [16,64] @ [64,128] + b2, out = x1 + z ---
#pragma unroll
  for (int t = 0; t < 2; ++t) {
    const int nt = wave + t * 4;  // 8 n-tiles of 16
    const float bias = b2[nt * 16 + nlane];
    v8f c;
#pragma unroll
    for (int e = 0; e < 8; ++e) c[e] = bias;
    for (int ks = 0; ks < MLPH_; ks += 32) {
      v16h a = pack_a_h(&ht[0][0] + ks, MLPH_);
      v16h b = pack_b_f32(w2 + ks * D_ + nt * 16, D_);
      c = wmma16(a, b, c);
    }
#pragma unroll
    for (int e = 0; e < 8; ++e) {
      const int m = e + hi * 8;
      const long off = (rowbase + m) * D_ + nt * 16 + nlane;
      out[off] = out[off] + c[e];   // out currently holds x1
    }
  }
}

// ---------------------------------------------------------------------------
// Launch. Workspace layout (needs 36 MB):
//   [0,8M)   qh f16 [B,H,S,HD]     [8M,16M)  kh      [16M,24M) vh
//   [24M,32M) x1h f16 [B*S,D]      [32M,34M) m_ws f32 [H,S,S]  [34M,36M) l_ws
// ---------------------------------------------------------------------------
extern "C" void kernel_launch(void* const* d_in, const int* in_sizes, int n_in,
                              void* d_out, int out_size, void* d_ws, size_t ws_size,
                              hipStream_t stream) {
  const float* x   = (const float*)d_in[0];
  const float* wq  = (const float*)d_in[1];
  const float* bq  = (const float*)d_in[2];
  const float* wk  = (const float*)d_in[3];
  const float* bk  = (const float*)d_in[4];
  const float* wv  = (const float*)d_in[5];
  const float* bv  = (const float*)d_in[6];
  const float* g1  = (const float*)d_in[7];
  const float* be1 = (const float*)d_in[8];
  // d_in[9]=g2, d_in[10]=be2: LN2 output is discarded in the reference.
  const float* w1  = (const float*)d_in[11];
  const float* b1  = (const float*)d_in[12];
  const float* w2  = (const float*)d_in[13];
  const float* b2  = (const float*)d_in[14];
  float* out = (float*)d_out;

  char* ws = (char*)d_ws;
  _Float16* qh  = (_Float16*)(ws);
  _Float16* kh  = (_Float16*)(ws + ((size_t)8 << 20));
  _Float16* vh  = (_Float16*)(ws + ((size_t)16 << 20));
  _Float16* x1h = (_Float16*)(ws + ((size_t)24 << 20));
  float*    m_ws = (float*)(ws + ((size_t)32 << 20));
  float*    l_ws = (float*)(ws + ((size_t)34 << 20));

  ln_qkv_kernel<<<dim3((B_ * S_) / 16), dim3(128), 0, stream>>>(
      x, wq, bq, wk, bk, wv, bv, g1, be1, qh, kh, vh);
  attn_stats_kernel<<<dim3(H_ * 16 * 16), dim3(32), 0, stream>>>(qh, kh, m_ws, l_ws);
  attn_apply_kernel<<<dim3(B_ * H_ * 16), dim3(32), 0, stream>>>(
      qh, kh, vh, m_ws, l_ws, x, out, x1h);
  mlp_kernel<<<dim3((B_ * S_) / 16), dim3(128), 0, stream>>>(x1h, w1, b1, w2, b2, out);
}
